// DiConvNeXtV2_8641474200446
// MI455X (gfx1250) — matrix operand order repaired and checked by reading, not compiled, not run on hardware
//
#include <hip/hip_runtime.h>
#include <math.h>

// ---- problem constants ----
#define CB   2
#define CC   192
#define CS   (48*48*48)      // 110592 spatial points per batch
#define CN   (CB*CS)         // 221184 total rows
#define C4   768

// LDS tile row stride in dwords: 20 = 16 pairs + 4 pad (keeps 16B alignment,
// breaks power-of-two bank pattern)
#define LDW  20

// ---- types for WMMA ----
typedef __attribute__((ext_vector_type(16))) __bf16        v16bf;
typedef __attribute__((ext_vector_type(8)))  float         v8f;
typedef __attribute__((ext_vector_type(8)))  unsigned int  v8u;
typedef __attribute__((ext_vector_type(4)))  unsigned int  v4u;

union FragAB { struct { v4u lo, hi; } q; v8u u; v16bf b; };

// double-buffered A/B tiles overlapped with the f32 staging buffer
union __align__(16) SU2 {
    struct { unsigned int A[2][128*LDW]; unsigned int Bt[2][64*LDW]; } s;  // 30.7 KB
    float stage[128*65];                                                   // 33.3 KB
};

// ---- bf16 helpers ----
__device__ __forceinline__ unsigned short f2bf(float f) {
    unsigned int u = __builtin_bit_cast(unsigned int, f);
    u += 0x7FFFu + ((u >> 16) & 1u);           // round-to-nearest-even
    return (unsigned short)(u >> 16);
}
__device__ __forceinline__ unsigned int pack2(float lo, float hi) {
    return ((unsigned int)f2bf(hi) << 16) | (unsigned int)f2bf(lo);
}

// low 32 bits of a generic shared pointer == LDS byte offset (flat aperture)
__device__ __forceinline__ unsigned int lds_off(const void* p) {
    return (unsigned int)(unsigned long long)p;
}

// CDNA5 async global->LDS copy (ASYNCcnt-tracked, no VGPR round trip).
// LDS destination address is per-lane independent of the global source ->
// free transpose while copying.
__device__ __forceinline__ void async_b32(unsigned int lds_byte_off,
                                          const unsigned int* sbase,
                                          unsigned int byte_off) {
    asm volatile("global_load_async_to_lds_b32 %0, %1, %2"
                 :
                 : "v"(lds_byte_off), "v"(byte_off), "s"(sbase)
                 : "memory");
}
__device__ __forceinline__ void wait_async0() {
    asm volatile("s_wait_asynccnt 0" ::: "memory");
}

// gather A/B fragments: row-major pair tiles -> two ds_load_b128 per fragment
__device__ __forceinline__ void gather_frags(const unsigned int* __restrict__ A,
                                             const unsigned int* __restrict__ Bt,
                                             int wm, int wn, int m16, int hi,
                                             FragAB a[2], FragAB bb[2]) {
    int ab    = hi ? 4 : 0;   // A 16x32 bf16: kp(v) = (v>=4?8:0) + 4*hi + (v&3)
    int bbase = hi ? 4 : 0;   // B 32x16 bf16: kp(v) = (v>=4?8:0) + 4*hi + (v&3)
#pragma unroll
    for (int t2 = 0; t2 < 2; ++t2) {
        int m = wm*32 + t2*16 + m16;
        a[t2].q.lo = *(const v4u*)(A + m*LDW + ab);
        a[t2].q.hi = *(const v4u*)(A + m*LDW + ab + 8);
        int j = wn*32 + t2*16 + m16;
        bb[t2].q.lo = *(const v4u*)(Bt + j*LDW + bbase);
        bb[t2].q.hi = *(const v4u*)(Bt + j*LDW + bbase + 8);
    }
}

// issue one 128x32 A tile + 32x64 B tile as async transposing copies
__device__ __forceinline__ void issue_tile(const unsigned int* Ab, const unsigned int* Bb,
                                           unsigned int ldsA, unsigned int ldsB,
                                           int kp0, int n0, int j0, int bstride, int tid) {
    int m = tid & 127, kbA = tid >> 7;
#pragma unroll
    for (int i = 0; i < 8; ++i) {
        int kp = kbA + 2*i;
        async_b32(ldsA + 4u*(unsigned)(m*LDW + kp), Ab,
                  4u*(unsigned)((kp0 + kp)*CN + n0 + m));
    }
    int j = tid & 63, kbB = tid >> 6;
#pragma unroll
    for (int i = 0; i < 4; ++i) {
        int kp = kbB + 4*i;
        async_b32(ldsB + 4u*(unsigned)(j*LDW + kp), Bb,
                  4u*(unsigned)((kp0 + kp)*bstride + j0 + j));
    }
}

// shared double-buffered async WMMA mainloop for both GEMMs
__device__ __forceinline__ void gemm_mainloop(SU2& sh, const unsigned int* Ab,
                                              const unsigned int* Bb, int iters,
                                              int n0, int j0, int bstride, int tid,
                                              v8f acc[2][2]) {
    int l = tid & 31, wid = tid >> 5;
    int wm = wid & 3, wn = wid >> 2;
    int m16 = l & 15, hi = l >> 4;

    unsigned int ldsA[2] = { lds_off(&sh.s.A[0][0]),  lds_off(&sh.s.A[1][0]) };
    unsigned int ldsB[2] = { lds_off(&sh.s.Bt[0][0]), lds_off(&sh.s.Bt[1][0]) };

    issue_tile(Ab, Bb, ldsA[0], ldsB[0], 0, n0, j0, bstride, tid);

    for (int it = 0; it < iters; ++it) {
        int cur = it & 1;
        wait_async0();          // current tile's DMA done (next not yet issued)
        __syncthreads();        // visible to all waves; prev reads finished
        if (it + 1 < iters)
            issue_tile(Ab, Bb, ldsA[cur ^ 1], ldsB[cur ^ 1],
                       (it + 1)*16, n0, j0, bstride, tid);
        FragAB a[2], bb[2];
        gather_frags(sh.s.A[cur], sh.s.Bt[cur], wm, wn, m16, hi, a, bb);
#pragma unroll
        for (int i = 0; i < 2; ++i)
#pragma unroll
            for (int j2 = 0; j2 < 2; ++j2)
                acc[i][j2] = __builtin_amdgcn_wmma_f32_16x16x32_bf16(
                    false, a[i].b, false, bb[j2].b, (short)0, acc[i][j2], false, false);
    }
    __syncthreads();            // before the union's stage buffer is reused
}

// ===================================================================
// Kernel 0: pair-pack w1 to bf16 + zero GRN accumulator
// ===================================================================
__global__ __launch_bounds__(256)
void k_prepw(const float* __restrict__ w1, unsigned int* __restrict__ w1p,
             float* __restrict__ gx2) {
    int t = blockIdx.x * 256 + threadIdx.x;
    if (t < 96*768) {             // w1p[kp][j] = pack(w1[2kp][j], w1[2kp+1][j])
        int kp = t / 768, j = t % 768;
        w1p[t] = pack2(w1[(2*kp)*C4 + j], w1[(2*kp+1)*C4 + j]);
    }
    if (t < CB*C4) gx2[t] = 0.0f;
}

// ===================================================================
// Kernel 1: fused depthwise 7x7x7 conv + bias + LayerNorm -> bf16 pairs
// one block = 4x4x8 spatial tile, all 192 channels, dynamic LDS
// ===================================================================
__global__ __launch_bounds__(128)
void k_convln(const float* __restrict__ x, const float* __restrict__ cw,
              const float* __restrict__ cb, const float* __restrict__ lng,
              const float* __restrict__ lnb, unsigned int* __restrict__ yn2) {
    extern __shared__ float dynsm[];
    float* xt = dynsm;            // 10*10*14 = 1400 input tile
    float* wS = dynsm + 1400;     // 343 weights (+1 pad)
    float* rS = dynsm + 1744;     // 192*128 per-channel conv results

    int bid = blockIdx.x;
    int xtile = bid % 6;
    int ytile = (bid / 6) % 12;
    int ztile = (bid / 72) % 12;
    int b     = bid / 864;
    int x0 = xtile * 8, y0 = ytile * 4, z0 = ztile * 4;

    int tid = threadIdx.x;
    int xx = tid & 7, yy = (tid >> 3) & 3, zz = tid >> 5;

    for (int c = 0; c < CC; ++c) {
        for (int i = tid; i < 343; i += 128) wS[i] = cw[i*CC + c];
        const float* xb = x + ((size_t)(b*CC + c)) * CS;
        for (int i = tid; i < 1400; i += 128) {
            int iz = i / 140; int r = i - iz*140;
            int iy = r / 14;  int ix = r - iy*14;
            int gz = z0 + iz - 3, gy = y0 + iy - 3, gx = x0 + ix - 3;
            float v = 0.0f;
            if ((unsigned)gz < 48u && (unsigned)gy < 48u && (unsigned)gx < 48u)
                v = xb[(gz*48 + gy)*48 + gx];
            xt[i] = v;
        }
        __syncthreads();
        float acc = cb[c];
        for (int kd = 0; kd < 7; ++kd)
            for (int kh = 0; kh < 7; ++kh) {
#pragma unroll
                for (int kw = 0; kw < 7; ++kw)
                    acc += wS[(kd*7 + kh)*7 + kw] *
                           xt[((zz + kd)*10 + (yy + kh))*14 + (xx + kw)];
            }
        rS[c*128 + tid] = acc;
        __syncthreads();
    }

    // LayerNorm over 192 channels for this thread's point
    float sum = 0.0f, sq = 0.0f;
    for (int c = 0; c < CC; ++c) { float v = rS[c*128 + tid]; sum += v; sq += v*v; }
    float mu   = sum * (1.0f / CC);
    float var  = sq  * (1.0f / CC) - mu*mu;
    float rstd = rsqrtf(var + 1e-6f);

    int n = b*CS + ((z0 + zz)*48 + (y0 + yy))*48 + (x0 + xx);
    for (int cp = 0; cp < 96; ++cp) {
        float v0 = (rS[(2*cp  )*128 + tid] - mu) * rstd * lng[2*cp  ] + lnb[2*cp  ];
        float v1 = (rS[(2*cp+1)*128 + tid] - mu) * rstd * lng[2*cp+1] + lnb[2*cp+1];
        yn2[(size_t)cp*CN + n] = pack2(v0, v1);
    }
}

// ===================================================================
// Kernel 2: GEMM1 (yn @ w1 + b1) -> exact GELU -> h (bf16 pairs)
//           + GRN sum-of-squares atomics. Block tile 128x64, WMMA bf16.
//           grid = (j-tiles, n-tiles): column blocks sharing an A tile are
//           adjacent in launch order -> A tile stays L2-hot.
// ===================================================================
__global__ __launch_bounds__(256)
void k_gemm1(const unsigned int* __restrict__ yn2, const unsigned int* __restrict__ w1p,
             const float* __restrict__ b1, unsigned int* __restrict__ h2,
             float* __restrict__ gx2) {
    __shared__ SU2 sh;
    __shared__ float red[256];

    int tid = threadIdx.x;
    int l = tid & 31, wid = tid >> 5;
    int wm = wid & 3, wn = wid >> 2;
    int m16 = l & 15, hi = l >> 4;
    int j0 = blockIdx.x * 64;
    int n0 = blockIdx.y * 128;

    v8f acc[2][2] = {};
    gemm_mainloop(sh, yn2, w1p, 6, n0, j0, C4, tid, acc);

    // epilogue: bias + exact GELU, stage f32 tile
#pragma unroll
    for (int j2 = 0; j2 < 2; ++j2) {
        int colL = wn*32 + j2*16 + m16;
        float bv = b1[j0 + colL];
#pragma unroll
        for (int i = 0; i < 2; ++i) {
            int mbase = wm*32 + i*16 + (hi ? 8 : 0);
#pragma unroll
            for (int r = 0; r < 8; ++r) {
                float v = acc[i][j2][r] + bv;
                v = 0.5f * v * (1.0f + erff(v * 0.70710678118f));
                sh.stage[(mbase + r)*65 + colL] = v;
            }
        }
    }
    __syncthreads();

    {   // pack h2 (bf16 pairs), coalesced along N
        int m = tid & 127, jb = tid >> 7;
#pragma unroll
        for (int i = 0; i < 16; ++i) {
            int jp = jb + 2*i;     // 0..31
            float v0 = sh.stage[m*65 + 2*jp];
            float v1 = sh.stage[m*65 + 2*jp + 1];
            h2[(size_t)(j0/2 + jp)*CN + n0 + m] = pack2(v0, v1);
        }
    }
    {   // GRN sum-of-squares per (batch, channel)
        int jl = tid & 63, mc = tid >> 6;
        float p = 0.0f;
        for (int m = mc*32; m < mc*32 + 32; ++m) {
            float v = sh.stage[m*65 + jl];
            p += v*v;
        }
        red[tid] = p;
        __syncthreads();
        if (tid < 64) {
            float tot = red[tid] + red[tid+64] + red[tid+128] + red[tid+192];
            int b = (n0 >= CS) ? 1 : 0;
            atomicAdd(&gx2[b*C4 + j0 + jl], tot);
        }
    }
}

// ===================================================================
// Kernel 3: GRN scale s = gamma*nx + 1, and fold beta through w2 + b2
// ===================================================================
__global__ __launch_bounds__(768)
void k_prep2(const float* __restrict__ gx2, const float* __restrict__ gamma,
             const float* __restrict__ beta, const float* __restrict__ w2,
             const float* __restrict__ b2, float* __restrict__ sarr,
             float* __restrict__ bias2) {
    __shared__ float sm[C4];
    int t = threadIdx.x;
    for (int b = 0; b < CB; ++b) {
        float g = sqrtf(gx2[b*C4 + t]);
        sm[t] = g;
        for (int st = 384; st > 0; st >>= 1) {
            __syncthreads();
            if (t < st) sm[t] += sm[t + st];
        }
        __syncthreads();
        float mean = sm[0] * (1.0f / C4);
        float nx = g / (mean + 1e-6f);
        sarr[b*C4 + t] = gamma[t]*nx + 1.0f;
        __syncthreads();
    }
    if (t < CC) {
        float acc = b2[t];
        for (int j = 0; j < C4; ++j) acc += beta[j] * w2[j*CC + t];
        bias2[t] = acc;
    }
}

// ===================================================================
// Kernel 3b: batch-scaled bf16 pair-pack of w2:  w2ps[b] = diag(s_b) * w2
// ===================================================================
__global__ __launch_bounds__(256)
void k_prepw2s(const float* __restrict__ w2, const float* __restrict__ sarr,
               unsigned int* __restrict__ w2ps) {
    int t = blockIdx.x * 256 + threadIdx.x;
    if (t < CB*384*192) {
        int b = t / (384*192); int r = t % (384*192);
        int kp = r / 192, c = r % 192;
        float s0 = sarr[b*C4 + 2*kp], s1 = sarr[b*C4 + 2*kp + 1];
        w2ps[t] = pack2(w2[(2*kp)*CC + c]*s0, w2[(2*kp+1)*CC + c]*s1);
    }
}

// ===================================================================
// Kernel 4: GEMM2 (h @ (s*w2) + bias2) + residual, NCDHW output
// ===================================================================
__global__ __launch_bounds__(256)
void k_gemm2(const unsigned int* __restrict__ h2, const unsigned int* __restrict__ w2ps,
             const float* __restrict__ bias2, const float* __restrict__ x,
             float* __restrict__ out) {
    __shared__ SU2 sh;

    int tid = threadIdx.x;
    int l = tid & 31, wid = tid >> 5;
    int wm = wid & 3, wn = wid >> 2;
    int m16 = l & 15, hi = l >> 4;
    int j0 = blockIdx.x * 64;
    int n0 = blockIdx.y * 128;
    int b = (n0 >= CS) ? 1 : 0;

    const unsigned int* Bb = w2ps + (size_t)b * (384*192);

    v8f acc[2][2] = {};
    gemm_mainloop(sh, h2, Bb, 24, n0, j0, CC, tid, acc);

    // stage f32 result + bias2
#pragma unroll
    for (int j2 = 0; j2 < 2; ++j2) {
        int colL = wn*32 + j2*16 + m16;
        float bv = bias2[j0 + colL];
#pragma unroll
        for (int i = 0; i < 2; ++i) {
            int mbase = wm*32 + i*16 + (hi ? 8 : 0);
#pragma unroll
            for (int r = 0; r < 8; ++r)
                sh.stage[(mbase + r)*65 + colL] = acc[i][j2][r] + bv;
        }
    }
    __syncthreads();

    // transpose-write to NCDHW + residual, coalesced along spatial dim
    int m = tid & 127, cbk = tid >> 7;
    int p = n0 + m - b*CS;
#pragma unroll
    for (int i = 0; i < 32; ++i) {
        int cl = cbk + 2*i;
        int c = j0 + cl;
        size_t idx = (size_t)(b*CC + c)*CS + p;
        out[idx] = sh.stage[m*65 + cl] + x[idx];
    }
}

// ===================================================================
extern "C" void kernel_launch(void* const* d_in, const int* in_sizes, int n_in,
                              void* d_out, int out_size, void* d_ws, size_t ws_size,
                              hipStream_t stream) {
    (void)in_sizes; (void)n_in; (void)out_size; (void)ws_size;

    const float* x      = (const float*)d_in[0];
    const float* conv_w = (const float*)d_in[1];
    const float* conv_b = (const float*)d_in[2];
    const float* ln_g   = (const float*)d_in[3];
    const float* ln_b   = (const float*)d_in[4];
    const float* w1     = (const float*)d_in[5];
    const float* b1     = (const float*)d_in[6];
    const float* grn_g  = (const float*)d_in[7];
    const float* grn_b  = (const float*)d_in[8];
    const float* w2     = (const float*)d_in[9];
    const float* b2     = (const float*)d_in[10];
    float* out = (float*)d_out;

    char* ws = (char*)d_ws;
    size_t off = 0;
    auto alloc = [&](size_t bytes) -> char* {
        char* p = ws + off;
        off = (off + bytes + 255) & ~(size_t)255;
        return p;
    };
    unsigned int* yn2   = (unsigned int*)alloc((size_t)96  * CN * 4);
    unsigned int* h2    = (unsigned int*)alloc((size_t)384 * CN * 4);
    unsigned int* w1p   = (unsigned int*)alloc((size_t)96  * C4 * 4);
    unsigned int* w2ps  = (unsigned int*)alloc((size_t)CB * 384 * CC * 4);
    float*        gx2   = (float*)alloc((size_t)CB * C4 * 4);
    float*        sarr  = (float*)alloc((size_t)CB * C4 * 4);
    float*        bias2 = (float*)alloc((size_t)CC * 4);

    // 0) weight packing + zero GRN accumulator
    k_prepw<<<288, 256, 0, stream>>>(w1, w1p, gx2);

    // 1) fused depthwise conv + LayerNorm -> yn2 (bf16 pairs, [96][N])
    size_t convln_lds = (size_t)(1400 + 344 + 192*128) * sizeof(float);
    k_convln<<<1728, 128, convln_lds, stream>>>(x, conv_w, conv_b, ln_g, ln_b, yn2);

    // 2) GEMM1 + GELU -> h2, GRN sumsq -> gx2   (grid: j fastest for L2 reuse)
    k_gemm1<<<dim3(12, 1728), 256, 0, stream>>>(yn2, w1p, b1, h2, gx2);

    // 3) GRN scale + folded bias, then batch-scaled w2 pack
    k_prep2<<<1, 768, 0, stream>>>(gx2, grn_g, grn_b, w2, b2, sarr, bias2);
    k_prepw2s<<<576, 256, 0, stream>>>(w2, sarr, w2ps);

    // 4) GEMM2 + residual -> out (NCDHW)   (grid: j fastest for L2 reuse)
    k_gemm2<<<dim3(3, 1728), 256, 0, stream>>>(h2, w2ps, bias2, x, out);
}
